// FlowMRI_Net_7971459301873
// MI455X (gfx1250) — matrix operand order repaired and checked by reading, not compiled, not run on hardware
//
#include <hip/hip_runtime.h>
#include <hip/hip_bf16.h>
#include <math.h>

// ---------------------------------------------------------------------------
// FlowMRI CRNN forward for MI455X (gfx1250, wave32).
// Complex 3x3 convs -> implicit GEMM on v_wmma_f32_16x16x32_f16.
// K order: k = (c*3+dy)*8 + dx*2+part  (each tap group padded 6->8 halves so
// every aligned 8-half A-fragment run = 4 consecutive LDS dwords; the 2 pad
// halves are zeroed in the packed B matrix, so A's pad lanes are don't-care).
// ---------------------------------------------------------------------------

typedef __attribute__((ext_vector_type(16))) _Float16     v16h;
typedef __attribute__((ext_vector_type(8)))  float        v8f;
typedef __attribute__((ext_vector_type(4)))  unsigned int v4u;

union V16H { v16h v; _Float16 h[16]; unsigned int u[8]; v4u u4[2]; };

#define HW      96
#define NPIX    9216      // 96*96
#define NT_T    12        // time frames
#define NF      48        // hidden features
#define NV      4         // image channels
#define NCOIL   8
#define CFRAME  18432     // floats per complex frame (9216*2)
#define CSTRIDE 221184    // floats per channel of a [c][t][pix] c64 tensor

__device__ __forceinline__ float sigmoidf_(float x) { return 1.f / (1.f + expf(-x)); }

// --------------------------- utility kernels -------------------------------

__global__ void zero_f32(float* __restrict__ p, long n) {
  long i = (long)blockIdx.x * blockDim.x + threadIdx.x;
  if (i < n) p[i] = 0.f;
}

// Centered ortho DFT matrices: F[u,p] = (-1)^(u+p) e^{-2pi i up/96}/sqrt(96)
__global__ void init_dft(float* __restrict__ Fc, float* __restrict__ Fci) {
  int idx = blockIdx.x * blockDim.x + threadIdx.x;
  if (idx >= NPIX) return;
  int u = idx / HW, p = idx % HW;
  float ang = -2.f * 3.14159265358979f * (float)(u * p) / 96.f;
  float s = ((u + p) & 1) ? -0.1020620726f : 0.1020620726f; // 1/sqrt(96)
  float cr = s * cosf(ang), ci = s * sinf(ang);
  Fc [idx * 2] = cr;  Fc [idx * 2 + 1] = ci;
  Fci[idx * 2] = cr;  Fci[idx * 2 + 1] = -ci;               // conjugate
}

// Pack complex OIHW weights into f16 B[k][n] stored n-major (Bp[n*Kpad+k]).
// Kpad = cinPad*24;  k -> grp = k>>3 (= c*3+dy), o = k&7 (= dx*2+part, 6,7 pad).
// Columns n<nf: real out ([wr,-wi]); n>=nf: imag out ([wi,wr]).
__global__ void pack_weights(const float* __restrict__ w, _Float16* __restrict__ Bp,
                             int nf, int cin, int cinPad, int Npad) {
  int Kpad = cinPad * 24;
  long total = (long)Npad * Kpad;
  long idx = (long)blockIdx.x * blockDim.x + threadIdx.x;
  if (idx >= total) return;
  int n = (int)(idx / Kpad), k = (int)(idx % Kpad);
  int grp = k >> 3, o = k & 7;
  int c = grp / 3, dy = grp - c * 3, dx = o >> 1, part = o & 1;
  float val = 0.f;
  if (o < 6 && n < 2 * nf && c < cin) {
    int ic = (n >= nf) ? 1 : 0;
    int ch = ic ? n - nf : n;
    const float* wp = w + (((long)ch * cin + c) * 9 + dy * 3 + dx) * 2;
    float wr = wp[0], wi = wp[1];
    val = ic ? (part ? wr : wi) : (part ? -wi : wr);
  }
  Bp[idx] = (_Float16)val;
}

// ------------------------ WMMA implicit-GEMM conv --------------------------
// Block = 128 threads (4 wave32), tile = 4 rows x 16 cols of pixels.
// grid.x = 144 tiles/frame, grid.y = frames.
template <int NTILES>
__global__ __launch_bounds__(128)
void cconv3x3_wmma(const float* __restrict__ in, long inStrideC, long inFrameStride,
                   const _Float16* __restrict__ Bp, int cin, int cinPad, int nf,
                   const float* __restrict__ bias,
                   const float* __restrict__ addend, long addStrideC, long addFrameStride,
                   float* __restrict__ out, long outStrideC, long outFrameStride) {
  // one dword per pixel (re,im f16); +1 dword pad for the dx=3 don't-care read
  __shared__ _Float16 patch[48 * 6 * 18 * 2 + 2];

  const int tid  = threadIdx.x;
  const int wave = tid >> 5, lane = tid & 31;
  const int tx = blockIdx.x % 6, ty = blockIdx.x / 6;
  const int x0 = tx * 16, y0 = ty * 4;
  const float* inF  = in  + (long)blockIdx.y * inFrameStride;
  const float* addF = addend ? addend + (long)blockIdx.y * addFrameStride : nullptr;
  float*       outF = out + (long)blockIdx.y * outFrameStride;

  // stage reflect-padded patch as f16 (re,im packed per dword)
  const int total = cinPad * 108;        // 6*18 per channel
  for (int i = tid; i < total; i += 128) {
    int c = i / 108, r = i % 108, py = r / 18, px = r % 18;
    int gy = y0 - 1 + py; gy = gy < 0 ? -gy : (gy > 95 ? 190 - gy : gy);
    int gx = x0 - 1 + px; gx = gx < 0 ? -gx : (gx > 95 ? 190 - gx : gx);
    float re = 0.f, im = 0.f;
    if (c < cin) {
      const float* p = inF + (long)c * inStrideC + (gy * HW + gx) * 2;
      re = p[0]; im = p[1];
    }
    patch[i * 2]     = (_Float16)re;
    patch[i * 2 + 1] = (_Float16)im;
  }
  __syncthreads();

  v8f acc[NTILES];
#pragma unroll
  for (int nt = 0; nt < NTILES; ++nt) acc[nt] = (v8f){0, 0, 0, 0, 0, 0, 0, 0};

  const int lx      = lane & 15;          // A row (pixel x within tile)
  const int ly      = wave;               // pixel y within tile
  const int kHi     = (lane >> 4) << 3;   // C/D: M offset 0 or 8
  const int kbOff   = (lane >> 4) << 3;   // B: K offset 0 or 8
  const int colLane = lane & 15;
  const int Kpad    = cinPad * 24;
  const int ksteps  = Kpad >> 5;          // 32 K (=4 tap groups) per step
  const unsigned int* pw = (const unsigned int*)patch;

  for (int ks = 0; ks < ksteps; ++ks) {
    const int kbase = ks << 5;
    // A fragment: run0 = tap group gA, run1 = gA+2; each run = 4 LDS dwords.
    int gA = (ks << 2) + (lane >> 4);
    int c0 = gA / 3, d0 = gA - c0 * 3;
    int gB = gA + 2;
    int c1 = gB / 3, d1 = gB - c1 * 3;
    int b0 = (c0 * 6 + ly + d0) * 18 + lx;
    int b1 = (c1 * 6 + ly + d1) * 18 + lx;
    V16H a;
#pragma unroll
    for (int j = 0; j < 4; ++j) {
      a.u[j]     = pw[b0 + j];
      a.u[4 + j] = pw[b1 + j];
    }
#pragma unroll
    for (int nt = 0; nt < NTILES; ++nt) {
      const _Float16* bp = Bp + (long)(nt * 16 + colLane) * Kpad + kbase + kbOff;
      if (nt == 0) __builtin_prefetch(bp + 32, 0, 1);   // next K-step's B
      V16H b;
      const v4u* bu = (const v4u*)bp;      // 16B-aligned: Kpad%32==0, offsets %8==0
      b.u4[0] = bu[0];
      b.u4[1] = bu[2];
      acc[nt] = __builtin_amdgcn_wmma_f32_16x16x32_f16(
          false, a.v, false, b.v, (short)0, acc[nt], false, false);
    }
  }

  // epilogue: C/D layout VGPR r -> (M = r + 8*(lane>=16), N = lane&15)
#pragma unroll
  for (int nt = 0; nt < NTILES; ++nt) {
    int n = nt * 16 + colLane;
    if (n < 2 * nf) {
      int part = (n >= nf) ? 1 : 0;
      int ch   = part ? n - nf : n;
      float bv = bias[ch * 2 + part];
#pragma unroll
      for (int r = 0; r < 8; ++r) {
        int m   = r + kHi;
        int pix = (y0 + ly) * HW + x0 + m;
        float val = acc[nt][r] + bv;
        if (addF) val += addF[(long)ch * addStrideC + pix * 2 + part];
        outF[(long)ch * outStrideC + pix * 2 + part] = val;
      }
    }
  }
}

// ------------------------ scan activation (modReLU) ------------------------
__global__ void modrelu_scan(const float* __restrict__ tmp, const float* __restrict__ bact,
                             float* __restrict__ h, float* __restrict__ outT,
                             long outStrideC, int add) {
  int idx = blockIdx.x * blockDim.x + threadIdx.x;
  if (idx >= NF * NPIX) return;
  int c = idx / NPIX, pix = idx % NPIX;
  float re = tmp[(long)c * CFRAME + pix * 2];
  float im = tmp[(long)c * CFRAME + pix * 2 + 1];
  float mag = sqrtf(re * re + im * im);
  float s = mag + bact[c];
  s = (s > 0.f ? s : 0.f) / (mag + 1e-8f);
  float hr = re * s, hi = im * s;
  h[(long)c * CFRAME + pix * 2]     = hr;
  h[(long)c * CFRAME + pix * 2 + 1] = hi;
  float* o = outT + (long)c * outStrideC + pix * 2;
  if (add) { o[0] += hr; o[1] += hi; }
  else     { o[0]  = hr; o[1]  = hi; }
}

// ------------------------ data-consistency kernels -------------------------
__global__ void dc_xs(const float* __restrict__ x, const float* __restrict__ cs,
                      float* __restrict__ xs) {
  long n = (long)NCOIL * NV * NT_T * NPIX;
  long idx = (long)blockIdx.x * blockDim.x + threadIdx.x;
  if (idx >= n) return;
  long pix = idx % NPIX; long r = idx / NPIX;
  int t = (int)(r % NT_T); r /= NT_T;
  int v = (int)(r % NV);
  int coil = (int)(r / NV);
  float cr = cs[((long)coil * NPIX + pix) * 2];
  float ci = cs[((long)coil * NPIX + pix) * 2 + 1];
  float xr = x[(((long)v * NT_T + t) * NPIX + pix) * 2];
  float xi = x[(((long)v * NT_T + t) * NPIX + pix) * 2 + 1];
  xs[idx * 2]     = cr * xr - ci * xi;
  xs[idx * 2 + 1] = cr * xi + ci * xr;
}

__global__ __launch_bounds__(96)
void dft_rows(const float* __restrict__ in, const float* __restrict__ F,
              float* __restrict__ out, int nimg) {
  int img = blockIdx.x / HW, u = blockIdx.x % HW, q = threadIdx.x;
  const float* ib = in + (long)img * CFRAME;
  float ar = 0.f, ai = 0.f;
  for (int p = 0; p < HW; ++p) {
    float fr = F[(u * HW + p) * 2], fi = F[(u * HW + p) * 2 + 1];
    float xr = ib[(p * HW + q) * 2], xi = ib[(p * HW + q) * 2 + 1];
    ar += fr * xr - fi * xi;
    ai += fr * xi + fi * xr;
  }
  out[(long)img * CFRAME + (u * HW + q) * 2]     = ar;
  out[(long)img * CFRAME + (u * HW + q) * 2 + 1] = ai;
}

__global__ __launch_bounds__(96)
void dft_cols(const float* __restrict__ in, const float* __restrict__ F,
              float* __restrict__ out, int nimg) {
  int img = blockIdx.x / HW, p = blockIdx.x % HW, v = threadIdx.x;
  const float* ib = in + (long)img * CFRAME;
  float ar = 0.f, ai = 0.f;
  for (int q = 0; q < HW; ++q) {
    float fr = F[(v * HW + q) * 2], fi = F[(v * HW + q) * 2 + 1];
    float xr = ib[(p * HW + q) * 2], xi = ib[(p * HW + q) * 2 + 1];
    ar += fr * xr - fi * xi;
    ai += fr * xi + fi * xr;
  }
  out[(long)img * CFRAME + (p * HW + v) * 2]     = ar;
  out[(long)img * CFRAME + (p * HW + v) * 2 + 1] = ai;
}

__global__ void dc_combine(float* __restrict__ kk, const float* __restrict__ kin,
                           const float* __restrict__ nl, int stage) {
  long n = (long)NCOIL * NV * NT_T * NPIX;
  long idx = (long)blockIdx.x * blockDim.x + threadIdx.x;
  if (idx >= n) return;
  long pix = idx % NPIX; long r = idx / NPIX;
  int t = (int)(r % NT_T); r /= NT_T;
  int v = (int)(r % NV);
  int coil = (int)(r / NV);
  long kidx = ((((long)v * NCOIL + coil) * NT_T + t) * NPIX + pix) * 2; // k0 = swapaxes(k,1,2)
  float k0r = kin[kidx], k0i = kin[kidx + 1];
  if (k0r != 0.f || k0i != 0.f) {
    float vs = sigmoidf_(nl[stage]);
    kk[idx * 2]     = vs * kk[idx * 2]     + (1.f - vs) * k0r;
    kk[idx * 2 + 1] = vs * kk[idx * 2 + 1] + (1.f - vs) * k0i;
  }
}

__global__ void dc_sum(const float* __restrict__ xi, const float* __restrict__ cs,
                       float* __restrict__ sx) {
  long n = (long)NV * NT_T * NPIX;
  long idx = (long)blockIdx.x * blockDim.x + threadIdx.x;
  if (idx >= n) return;
  long pix = idx % NPIX; long r = idx / NPIX;
  int t = (int)(r % NT_T);
  int v = (int)(r / NT_T);
  float sr = 0.f, si = 0.f;
  for (int coil = 0; coil < NCOIL; ++coil) {
    long xidx = ((((long)coil * NV + v) * NT_T + t) * NPIX + pix) * 2;
    float ar = xi[xidx], ai = xi[xidx + 1];
    float cr = cs[((long)coil * NPIX + pix) * 2];
    float ci = cs[((long)coil * NPIX + pix) * 2 + 1];
    sr += ar * cr + ai * ci;   // xi * conj(c)
    si += ai * cr - ar * ci;
  }
  sx[idx * 2] = sr;
  sx[idx * 2 + 1] = si;
}

__global__ void update_x(const float* __restrict__ x, const float* __restrict__ xcnn,
                         const float* __restrict__ sx, const float* __restrict__ para,
                         int stage, float* __restrict__ xo) {
  long n = (long)NV * NT_T * NPIX * 2;
  long idx = (long)blockIdx.x * blockDim.x + threadIdx.x;
  if (idx >= n) return;
  float p = sigmoidf_(para[stage]);
  xo[idx] = p * (x[idx] + xcnn[idx]) + (1.f - p) * sx[idx];
}

// --------------------------------- host ------------------------------------

extern "C" void kernel_launch(void* const* d_in, const int* in_sizes, int n_in,
                              void* d_out, int out_size, void* d_ws, size_t ws_size,
                              hipStream_t stream) {
  (void)in_sizes; (void)n_in; (void)out_size; (void)ws_size;

  const float* X   = (const float*)d_in[0];   // (1,4,12,1,96,96) c64
  const float* Kin = (const float*)d_in[1];   // (1,4,8,12,1,96,96) c64
  const float* Cs  = (const float*)d_in[2];   // (1,8,1,96,96) c64
  const float* Wl[4][7];
  for (int L = 0; L < 4; ++L)
    for (int j = 0; j < 7; ++j) Wl[L][j] = (const float*)d_in[3 + 7 * L + j];
  const float* Wc4  = (const float*)d_in[31];
  const float* Bc4  = (const float*)d_in[32];
  const float* NL   = (const float*)d_in[33];
  const float* PARA = (const float*)d_in[34];

  // workspace carve-out (256B aligned)
  unsigned long long base = (unsigned long long)d_ws;
  auto alloc = [&](size_t bytes) -> void* {
    void* p = (void*)base;
    base += (bytes + 255) & ~(size_t)255;
    return p;
  };
  float* Fc  = (float*)alloc((size_t)NPIX * 2 * 4);
  float* Fci = (float*)alloc((size_t)NPIX * 2 * 4);
  _Float16* Bi2h[4]; _Float16* Bh2h[4]; _Float16* Bih[4];
  for (int L = 0; L < 4; ++L) {
    int cinPad = L ? 48 : 16;
    Bi2h[L] = (_Float16*)alloc((size_t)96 * cinPad * 24 * 2);
    Bh2h[L] = (_Float16*)alloc((size_t)96 * 1152 * 2);
    Bih[L]  = (_Float16*)alloc((size_t)96 * 1152 * 2);
  }
  _Float16* Bc = (_Float16*)alloc((size_t)16 * 1152 * 2);

  const size_t HIDB = (size_t)NF * NT_T * NPIX * 2 * 4;  // 42.5 MB
  float* hid[4];
  for (int L = 0; L < 4; ++L) hid[L] = (float*)alloc(HIDB);
  float* scr  = (float*)alloc(HIDB);
  float* PRE  = (float*)alloc(HIDB);
  float* Htmp = (float*)alloc((size_t)NF * NPIX * 2 * 4);
  float* TMP  = (float*)alloc((size_t)NF * NPIX * 2 * 4);
  float* XBUF = (float*)alloc((size_t)NV * NT_T * NPIX * 2 * 4);
  float* XCNN = (float*)alloc((size_t)NV * NT_T * NPIX * 2 * 4);
  float* SX   = (float*)alloc((size_t)NV * NT_T * NPIX * 2 * 4);
  float* DA   = (float*)alloc((size_t)NCOIL * NV * NT_T * NPIX * 2 * 4);
  float* DB   = (float*)alloc((size_t)NCOIL * NV * NT_T * NPIX * 2 * 4);

  auto zero = [&](float* p, long n) {
    zero_f32<<<(unsigned)((n + 255) / 256), 256, 0, stream>>>(p, n);
  };
  auto pack = [&](const float* w, _Float16* bp, int nf, int cin, int cinPad, int Npad) {
    long total = (long)Npad * cinPad * 24;
    pack_weights<<<(unsigned)((total + 255) / 256), 256, 0, stream>>>(w, bp, nf, cin, cinPad, Npad);
  };
  auto conv = [&](int frames, const float* in, long inSC, long inFS,
                  const _Float16* Bp, int cin, int cinPad, int nf, int ntiles,
                  const float* bias, const float* add, long addSC, long addFS,
                  float* out, long outSC, long outFS) {
    dim3 g(144, (unsigned)frames);
    if (ntiles == 6)
      cconv3x3_wmma<6><<<g, 128, 0, stream>>>(in, inSC, inFS, Bp, cin, cinPad, nf, bias,
                                              add, addSC, addFS, out, outSC, outFS);
    else
      cconv3x3_wmma<1><<<g, 128, 0, stream>>>(in, inSC, inFS, Bp, cin, cinPad, nf, bias,
                                              add, addSC, addFS, out, outSC, outFS);
  };

  // init: DFT matrices, packed weights, zero hidden state
  init_dft<<<(NPIX + 255) / 256, 256, 0, stream>>>(Fc, Fci);
  for (int L = 0; L < 4; ++L) {
    pack(Wl[L][0], Bi2h[L], NF, L ? NF : NV, L ? 48 : 16, 96);
    pack(Wl[L][2], Bh2h[L], NF, NF, 48, 96);
    pack(Wl[L][4], Bih[L],  NF, NF, 48, 96);
  }
  pack(Wc4, Bc, NV, NF, 48, 16);
  for (int L = 0; L < 4; ++L) zero(hid[L], (long)NF * NT_T * NPIX * 2);

  auto bcrnn = [&](const float* inp, int cin, const float* itr, int L, float* OUT) {
    int cinPad = (cin == NV) ? 16 : 48;
    // pre = cconv(inp, w_i2h)+b_i2h + cconv(itr, w_ih2ih)+b_ih2ih, all 12 frames
    conv(NT_T, inp, CSTRIDE, CFRAME, Bi2h[L], cin, cinPad, NF, 6, Wl[L][1],
         nullptr, 0, 0, PRE, CSTRIDE, CFRAME);
    conv(NT_T, itr, CSTRIDE, CFRAME, Bih[L], NF, 48, NF, 6, Wl[L][5],
         PRE, CSTRIDE, CFRAME, PRE, CSTRIDE, CFRAME);
    // forward (dir=0, SET) and backward (dir=1, ADD) scans
    for (int dir = 0; dir < 2; ++dir) {
      zero(Htmp, (long)NF * NPIX * 2);
      for (int s = 0; s < NT_T; ++s) {
        int t = dir ? (NT_T - 1 - s) : s;
        conv(1, Htmp, CFRAME, 0, Bh2h[L], NF, 48, NF, 6, Wl[L][3],
             PRE + (long)t * CFRAME, CSTRIDE, 0, TMP, CFRAME, 0);
        modrelu_scan<<<(NF * NPIX + 255) / 256, 256, 0, stream>>>(
            TMP, Wl[L][6], Htmp, OUT + (long)t * CFRAME, CSTRIDE, dir);
      }
    }
  };

  const int nimg = NCOIL * NV * NT_T;  // 384 images per FFT pass
  for (int i = 0; i < 2; ++i) {
    const float* xcur = i ? XBUF : X;
    float* xout = i ? (float*)d_out : XBUF;

    for (int L = 0; L < 4; ++L) {
      const float* inp = L ? hid[L - 1] : xcur;
      bcrnn(inp, L ? NF : NV, hid[L], L, scr);
      float* tp = hid[L]; hid[L] = scr; scr = tp;   // new hidden; recycle old
    }
    // conv4: nf=4 outputs (Npad=16, 1 N-tile)
    conv(NT_T, hid[3], CSTRIDE, CFRAME, Bc, NF, 48, NV, 1, Bc4,
         nullptr, 0, 0, XCNN, CSTRIDE, CFRAME);

    // data consistency
    long nks = (long)NCOIL * NV * NT_T * NPIX;
    dc_xs<<<(unsigned)((nks + 255) / 256), 256, 0, stream>>>(xcur, Cs, DA);
    dft_rows<<<nimg * HW, HW, 0, stream>>>(DA, Fc, DB, nimg);
    dft_cols<<<nimg * HW, HW, 0, stream>>>(DB, Fc, DA, nimg);
    dc_combine<<<(unsigned)((nks + 255) / 256), 256, 0, stream>>>(DA, Kin, NL, i);
    dft_rows<<<nimg * HW, HW, 0, stream>>>(DA, Fci, DB, nimg);
    dft_cols<<<nimg * HW, HW, 0, stream>>>(DB, Fci, DA, nimg);
    long nvx = (long)NV * NT_T * NPIX;
    dc_sum<<<(unsigned)((nvx + 255) / 256), 256, 0, stream>>>(DA, Cs, SX);
    update_x<<<(unsigned)((nvx * 2 + 255) / 256), 256, 0, stream>>>(
        xcur, XCNN, SX, PARA, i, xout);
  }
}
